// OptimizedSimEncoderLayer_37383395344629
// MI455X (gfx1250) — matrix operand order repaired and checked by reading, not compiled, not run on hardware
//
#include <hip/hip_runtime.h>

typedef __bf16 bf16_t;
typedef __attribute__((ext_vector_type(16))) __bf16 bfx16;
typedef __attribute__((ext_vector_type(8)))  __bf16 bfx8;
typedef __attribute__((ext_vector_type(8)))  float  fx8;

namespace {
constexpr int kB    = 16;
constexpr int kN    = 257;
constexpr int kNl   = 514;
constexpr int kNr   = 512;
constexpr int kD    = 1024;
constexpr int kH    = 16;
constexpr int kHd   = 64;
constexpr int kNpad = 288;    // 3 flash chunks of 96 keys (6 subtiles each)
constexpr int kQT   = 17;     // ceil(257/16)
constexpr int kRows = kB * kN;        // 4112
constexpr int kRowsPad = 4128;        // 129 * 32 (projection M padding)
constexpr float kScale = 0.125f;      // hd^-0.5
constexpr float kLog2e = 1.44269504088896340736f;
}

static __device__ __forceinline__ fx8 fzero8() {
  fx8 z = {0.f, 0.f, 0.f, 0.f, 0.f, 0.f, 0.f, 0.f};
  return z;
}

static __device__ __forceinline__ bfx16 cat8(bfx8 lo, bfx8 hi) {
  return __builtin_shufflevector(lo, hi, 0,1,2,3,4,5,6,7,8,9,10,11,12,13,14,15);
}

// ---------------------------------------------------------------------------
// Kernel 1: gather token rows + convert to WMMA-friendly bf16 layouts.
//   hid  : [4112, D]    f32  (residual term, gathered[0])
//   qh   : [B,H,288,64] bf16, q pre-scaled by hd^-0.5
//   kh   : [B,H,288,64] bf16
//   vth  : [B,H,64,288] bf16 (V transposed so PV B-fragments are contiguous)
// ---------------------------------------------------------------------------
__global__ void gather_rows_kernel(const int* __restrict__ gidx,
                                   const float* __restrict__ lhs,
                                   const float* __restrict__ rhs,
                                   float* __restrict__ hid,
                                   bf16_t* __restrict__ qh,
                                   bf16_t* __restrict__ kh,
                                   bf16_t* __restrict__ vth) {
  const int row = blockIdx.x;              // b*N + n
  const int b   = row / kN;
  const int n   = row - b * kN;
  const int idx = gidx[row];
  const int d0  = threadIdx.x * 4;

  const float* s0;
  size_t stride;
  if (idx < kB * kNl) {
    s0 = lhs + (size_t)idx * kD;
    stride = (size_t)kB * kNl * kD;
  } else {
    s0 = rhs + (size_t)(idx - kB * kNl) * kD;
    stride = (size_t)kNr * kD;
  }
  const float* s1 = s0 + stride;
  const float* s2 = s0 + 2 * stride;
  const float* s3 = s0 + 3 * stride;

  float4 hv = *(const float4*)(s0 + d0);
  *(float4*)(hid + (size_t)row * kD + d0) = hv;

  float4 qv = *(const float4*)(s1 + d0);
  float4 kv = *(const float4*)(s2 + d0);
  float4 vv = *(const float4*)(s3 + d0);

  const int h = d0 >> 6;
  const int c = d0 & 63;
  const size_t qko = ((size_t)(b * kH + h) * kNpad + n) * kHd + c;
  qh[qko + 0] = (bf16_t)(qv.x * kScale);
  qh[qko + 1] = (bf16_t)(qv.y * kScale);
  qh[qko + 2] = (bf16_t)(qv.z * kScale);
  qh[qko + 3] = (bf16_t)(qv.w * kScale);
  kh[qko + 0] = (bf16_t)kv.x;
  kh[qko + 1] = (bf16_t)kv.y;
  kh[qko + 2] = (bf16_t)kv.z;
  kh[qko + 3] = (bf16_t)kv.w;

  const size_t vo = ((size_t)(b * kH + h) * kHd + c) * kNpad + n;
  vth[vo + 0 * kNpad] = (bf16_t)vv.x;
  vth[vo + 1 * kNpad] = (bf16_t)vv.y;
  vth[vo + 2 * kNpad] = (bf16_t)vv.z;
  vth[vo + 3 * kNpad] = (bf16_t)vv.w;
}

// ---------------------------------------------------------------------------
// Kernel 2: convert Wo to bf16.
// ---------------------------------------------------------------------------
__global__ void wo_convert_kernel(const float* __restrict__ wo,
                                  bf16_t* __restrict__ wob) {
  const size_t i = ((size_t)blockIdx.x * blockDim.x + threadIdx.x) * 4;
  float4 w = *(const float4*)(wo + i);
  wob[i + 0] = (bf16_t)w.x;
  wob[i + 1] = (bf16_t)w.y;
  wob[i + 2] = (bf16_t)w.z;
  wob[i + 3] = (bf16_t)w.w;
}

// ---------------------------------------------------------------------------
// Kernel 3: flash attention. One wave32 per (query tile, head, batch).
// Grid (17, 16, 16), block 32.  Online softmax over 3 key-chunks of 96
// (wide chunks amortize the cross-lane reduction stages 3x vs 32-key chunks).
// ---------------------------------------------------------------------------
__global__ __launch_bounds__(32) void attn_kernel(const bf16_t* __restrict__ qh,
                                                  const bf16_t* __restrict__ kh,
                                                  const bf16_t* __restrict__ vth,
                                                  bf16_t* __restrict__ aout) {
  const int qt = blockIdx.x;
  const int h  = blockIdx.y;
  const int b  = blockIdx.z;
  const int l  = threadIdx.x;
  const int lm = l & 15;          // column lane / A-row lane
  const int hh = l >> 4;          // lane half
  const int h8 = hh * 8;

  const bf16_t* qbase = qh  + (size_t)(b * kH + h) * kNpad * kHd;
  const bf16_t* kbase = kh  + (size_t)(b * kH + h) * kNpad * kHd;
  const bf16_t* vbase = vth + (size_t)(b * kH + h) * kHd * kNpad;

  // Q A-fragments for this 16-row tile (K = 64 -> two 32-chunks).
  const int qrow = qt * 16 + lm;  // < 288; rows >= 257 are garbage but never stored
  bfx16 qf[2];
#pragma unroll
  for (int c = 0; c < 2; ++c) {
    qf[c] = *(const bfx16*)(qbase + (size_t)qrow * kHd + c * 32 + hh * 16);
  }

  fx8 o[4];
#pragma unroll
  for (int t = 0; t < 4; ++t) o[t] = fzero8();
  float mrun[8], lrun[8];
#pragma unroll
  for (int r = 0; r < 8; ++r) { mrun[r] = -3.0e38f; lrun[r] = 0.f; }

  __shared__ __align__(32) bf16_t Pst[16 * 96];   // probs staging (C -> A relayout)

  for (int ch = 0; ch < 3; ++ch) {
    // ---- scores: six 16-key subtiles, contraction over hd=64 ----
    fx8 s[6];
#pragma unroll
    for (int st = 0; st < 6; ++st) {
      const int key = ch * 96 + st * 16 + lm;
      bfx16 k0 = *(const bfx16*)(kbase + (size_t)key * kHd + hh * 16);
      bfx16 k1 = *(const bfx16*)(kbase + (size_t)key * kHd + 32 + hh * 16);
      fx8 acc = fzero8();
      acc = __builtin_amdgcn_wmma_f32_16x16x32_bf16(false, qf[0], false, k0,
                                                    (short)0, acc, false, false);
      acc = __builtin_amdgcn_wmma_f32_16x16x32_bf16(false, qf[1], false, k1,
                                                    (short)0, acc, false, false);
      if (key >= kN) {
#pragma unroll
        for (int r = 0; r < 8; ++r) acc[r] = -1.0e30f;
      }
      s[st] = acc;
    }

    // ---- online softmax per row (rows r + 8*hh live in this lane half) ----
#pragma unroll
    for (int r = 0; r < 8; ++r) {
      float mx = s[0][r];
#pragma unroll
      for (int st = 1; st < 6; ++st) mx = fmaxf(mx, s[st][r]);
      mx = fmaxf(mx, __shfl_xor(mx, 1, 32));
      mx = fmaxf(mx, __shfl_xor(mx, 2, 32));
      mx = fmaxf(mx, __shfl_xor(mx, 4, 32));
      mx = fmaxf(mx, __shfl_xor(mx, 8, 32));
      const float mnew  = fmaxf(mrun[r], mx);
      const float alpha = exp2f((mrun[r] - mnew) * kLog2e);
      float rs = 0.f;
#pragma unroll
      for (int st = 0; st < 6; ++st) {
        const float p = exp2f((s[st][r] - mnew) * kLog2e);
        rs += p;
        // C-layout (row = h8+r, col = st*16+lm) -> LDS [qrow][key]
        Pst[(h8 + r) * 96 + st * 16 + lm] = (bf16_t)p;
      }
      rs += __shfl_xor(rs, 1, 32);
      rs += __shfl_xor(rs, 2, 32);
      rs += __shfl_xor(rs, 4, 32);
      rs += __shfl_xor(rs, 8, 32);
      lrun[r] = lrun[r] * alpha + rs;
      mrun[r] = mnew;
#pragma unroll
      for (int t = 0; t < 4; ++t) o[t][r] *= alpha;
    }

    // make staged probs visible before re-reading as A fragments
    asm volatile("s_wait_dscnt 0x0" ::: "memory");

    // ---- P as A fragments (16 q x 32 keys), V^T as B fragments ----
#pragma unroll
    for (int g = 0; g < 3; ++g) {
      bfx16 pf = *(const bfx16*)(&Pst[lm * 96 + g * 32 + hh * 16]);
      asm volatile("" ::: "memory");
#pragma unroll
      for (int t = 0; t < 4; ++t) {
        bfx16 vf = *(const bfx16*)(vbase + (size_t)(t * 16 + lm) * kNpad +
                                   ch * 96 + g * 32 + hh * 16);
        o[t] = __builtin_amdgcn_wmma_f32_16x16x32_bf16(false, pf, false, vf,
                                                       (short)0, o[t], false, false);
      }
    }
  }

  // ---- epilogue: normalize, write bf16 attn_out [4112, D] ----
  const size_t obase = (size_t)b * kN * kD + (size_t)h * kHd;
#pragma unroll
  for (int t = 0; t < 4; ++t) {
#pragma unroll
    for (int r = 0; r < 8; ++r) {
      const int m = qt * 16 + h8 + r;
      if (m < kN)
        aout[obase + (size_t)m * kD + t * 16 + lm] = (bf16_t)(o[t][r] / lrun[r]);
    }
  }
}

// ---------------------------------------------------------------------------
// Kernel 4: out = attn_out @ Wo^T + bo + hidden.  One wave per (32 rows x 128
// cols): two M-tiles share every Wo B-fragment, halving L2 Wo re-reads.
// Grid (129, 8); 129*32 = 4128 rows (aout padded, stores guarded at 4112).
// ---------------------------------------------------------------------------
__global__ __launch_bounds__(32) void proj_kernel(const bf16_t* __restrict__ aout,
                                                  const bf16_t* __restrict__ wob,
                                                  const float* __restrict__ bo,
                                                  const float* __restrict__ hid,
                                                  float* __restrict__ out) {
  const int bm = blockIdx.x;      // 32-row M block
  const int ng = blockIdx.y;      // 128-col group
  const int l  = threadIdx.x;
  const int lm = l & 15;
  const int hh = l >> 4;
  const int h8 = hh * 8;
  const int mrow0 = bm * 32 + lm;        // M tile 0 row for A fragment
  const int mrow1 = bm * 32 + 16 + lm;   // M tile 1 row

  fx8 acc0[8], acc1[8];
#pragma unroll
  for (int t = 0; t < 8; ++t) { acc0[t] = fzero8(); acc1[t] = fzero8(); }

  for (int kc = 0; kc < 32; ++kc) {
    bfx16 af0 = *(const bfx16*)(aout + (size_t)mrow0 * kD + kc * 32 + hh * 16);
    bfx16 af1 = *(const bfx16*)(aout + (size_t)mrow1 * kD + kc * 32 + hh * 16);
#pragma unroll
    for (int t = 0; t < 8; ++t) {
      const int col = ng * 128 + t * 16 + lm;
      // B(k, col) = Wo[col][k] : contiguous in k, 32B per lane; reused by both M tiles
      bfx16 bfr = *(const bfx16*)(wob + (size_t)col * kD + kc * 32 + hh * 16);
      acc0[t] = __builtin_amdgcn_wmma_f32_16x16x32_bf16(false, af0, false, bfr,
                                                        (short)0, acc0[t], false, false);
      acc1[t] = __builtin_amdgcn_wmma_f32_16x16x32_bf16(false, af1, false, bfr,
                                                        (short)0, acc1[t], false, false);
    }
  }

#pragma unroll
  for (int t = 0; t < 8; ++t) {
    const int col = ng * 128 + t * 16 + lm;
    const float bias = bo[col];
#pragma unroll
    for (int r = 0; r < 8; ++r) {
      const size_t row0 = (size_t)bm * 32 + h8 + r;
      out[row0 * kD + col] = acc0[t][r] + bias + hid[row0 * kD + col];
      const size_t row1 = (size_t)bm * 32 + 16 + h8 + r;
      if (row1 < (size_t)kRows)
        out[row1 * kD + col] = acc1[t][r] + bias + hid[row1 * kD + col];
    }
  }
}

// ---------------------------------------------------------------------------
extern "C" void kernel_launch(void* const* d_in, const int* in_sizes, int n_in,
                              void* d_out, int out_size, void* d_ws, size_t ws_size,
                              hipStream_t stream) {
  (void)in_sizes; (void)n_in; (void)out_size; (void)ws_size;
  const int*   gidx = (const int*)d_in[0];
  const float* lhs  = (const float*)d_in[1];
  const float* rhs  = (const float*)d_in[2];
  const float* Wo   = (const float*)d_in[3];
  const float* bo   = (const float*)d_in[4];
  float* out = (float*)d_out;

  char* ws = (char*)d_ws;
  size_t off = 0;
  float*  hid = (float*)(ws + off);  off += (size_t)kRows * kD * sizeof(float);
  bf16_t* qh  = (bf16_t*)(ws + off); off += (size_t)kB * kH * kNpad * kHd * sizeof(bf16_t);
  bf16_t* kh  = (bf16_t*)(ws + off); off += (size_t)kB * kH * kNpad * kHd * sizeof(bf16_t);
  bf16_t* vth = (bf16_t*)(ws + off); off += (size_t)kB * kH * kHd * kNpad * sizeof(bf16_t);
  bf16_t* aout= (bf16_t*)(ws + off); off += (size_t)kRowsPad * kD * sizeof(bf16_t);
  bf16_t* wob = (bf16_t*)(ws + off); off += (size_t)kD * kD * sizeof(bf16_t);

  gather_rows_kernel<<<kRows, 256, 0, stream>>>(gidx, lhs, rhs, hid, qh, kh, vth);
  wo_convert_kernel<<<(kD * kD) / (256 * 4), 256, 0, stream>>>(Wo, wob);
  attn_kernel<<<dim3(kQT, kH, kB), 32, 0, stream>>>(qh, kh, vth, aout);
  proj_kernel<<<dim3(129, 8), 32, 0, stream>>>(aout, wob, bo, hid, out);
}
